// GATMultiHead_2894807957582
// MI455X (gfx1250) — matrix operand order repaired and checked by reading, not compiled, not run on hardware
//
#include <hip/hip_runtime.h>

#define N_NODES 50000
#define N_EDGES 1600000
#define ETOT    (N_EDGES + N_NODES)
#define HH      2
#define NGRAPH  64
#define EPSV    1e-5f
#define SLOPE   0.2f

typedef __attribute__((ext_vector_type(16))) _Float16 v16h;
typedef __attribute__((ext_vector_type(8)))  _Float16 v8h;
typedef __attribute__((ext_vector_type(8)))  float    v8f;

union U16h { v16h v; v8h h[2]; _Float16 e[16]; };
union U8f  { v8f v; float e[8]; };

// Monotone float <-> uint key for atomic segment-max.
static __device__ __forceinline__ unsigned fkey(float f) {
  unsigned u = __float_as_uint(f);
  return (u & 0x80000000u) ? ~u : (u | 0x80000000u);
}
static __device__ __forceinline__ float fdec(unsigned k) {
  unsigned u = (k & 0x80000000u) ? (k & 0x7FFFFFFFu) : ~k;
  return __uint_as_float(u);
}

static __device__ __forceinline__ void edge_nodes(const int* __restrict__ ei, int eid,
                                                  int& s, int& d) {
  if (eid < N_EDGES) { s = ei[eid]; d = ei[N_EDGES + eid]; }
  else               { s = eid - N_EDGES; d = s; }   // self loops
}

// -------- fp32 -> fp16 weight conversion WITH transpose: wt[j*K + k] = w[k*ncols + j] --------
// Makes the WMMA B fragment (two 8-half K runs per lane) two b128 loads.
__global__ void k_f2h_t(const float* __restrict__ w, _Float16* __restrict__ wt,
                        int K, int ncols) {
  int i = blockIdx.x * blockDim.x + threadIdx.x;
  if (i >= K * ncols) return;
  int k = i / ncols, j = i % ncols;
  wt[j * K + k] = (_Float16)w[i];
}

// ---------------- layer 1 linear: [N,3] @ [3,64] ----------------
__global__ void k_lin1(const float* __restrict__ x, const float* __restrict__ w1,
                       float* __restrict__ hlin) {
  int i = blockIdx.x * blockDim.x + threadIdx.x;
  if (i >= N_NODES * 64) return;
  int n = i >> 6, j = i & 63;
  float a = x[n * 5 + 0] * w1[j] + x[n * 5 + 1] * w1[64 + j] + x[n * 5 + 2] * w1[128 + j];
  hlin[i] = a;
}

// ---------------- WMMA GEMM: Out[N,ncols] = A[N,K](f16) @ Bt[ncols,K](f16, pre-transposed) ----
// One wave per 16x16 output tile; K in steps of 32 via v_wmma_f32_16x16x32_f16.
__global__ void k_gemm_wmma(const _Float16* __restrict__ A, const _Float16* __restrict__ Bt,
                            float* __restrict__ Out, int K, int ncols) {
  int wave = (blockIdx.x * blockDim.x + threadIdx.x) >> 5;
  int lane = threadIdx.x & 31;
  int col_tiles = ncols >> 4;
  int row = (wave / col_tiles) << 4;
  int col = (wave % col_tiles) << 4;
  if (row >= N_NODES) return;

  int mlane = lane & 15;
  int half  = lane >> 4;
  int r   = row + mlane;        // A row for this lane
  int kb  = half * 8;           // A K-base per lane half (ISA 16-bit A layout)
  int kb2 = half * 16;          // unused; B loads use kb, as A
  int nn  = col + mlane;        // B column for this lane

  v8f c = {};
  for (int ks = 0; ks < K; ks += 32) {
    U16h ua, ub;
    // A fragment: K = ks+kb..+7 (elems 0..7) and ks+16+kb..+7 (elems 8..15); 16B loads.
    ua.h[0] = *reinterpret_cast<const v8h*>(A + (size_t)r * K + ks + kb);
    ua.h[1] = *reinterpret_cast<const v8h*>(A + (size_t)r * K + ks + 16 + kb);
    // B fragment: column nn, K = ks+kb..+7 (elems 0..7) and ks+16+kb..+7 (elems 8..15), matching A.
    ub.h[0] = *reinterpret_cast<const v8h*>(Bt + (size_t)nn * K + ks + kb);
    ub.h[1] = *reinterpret_cast<const v8h*>(Bt + (size_t)nn * K + ks + 16 + kb);
    c = __builtin_amdgcn_wmma_f32_16x16x32_f16(false, ua.v, false, ub.v,
                                               (short)0, c, false, false);
  }
  U8f uc; uc.v = c;
#pragma unroll
  for (int vv = 0; vv < 8; ++vv) {
    int M = row + half * 8 + vv;
    Out[(size_t)M * ncols + (col + mlane)] = uc.e[vv];
  }
}

// ---------------- attention dots: a_s/a_d [N,H] ----------------
__global__ void k_attdot(const float* __restrict__ hlin, const float* __restrict__ atts,
                         const float* __restrict__ attd, float* __restrict__ a_s,
                         float* __restrict__ a_d, int C) {
  int i = blockIdx.x * blockDim.x + threadIdx.x;
  if (i >= N_NODES * HH) return;
  int n = i / HH, h = i % HH;
  const float4* hp = reinterpret_cast<const float4*>(hlin + (size_t)n * HH * C + h * C);
  const float4* sp = reinterpret_cast<const float4*>(atts + h * C);
  const float4* dp = reinterpret_cast<const float4*>(attd + h * C);
  float ss = 0.f, sd = 0.f;
  for (int c = 0; c < (C >> 2); ++c) {
    float4 v = hp[c], s4 = sp[c], d4 = dp[c];
    ss += v.x * s4.x + v.y * s4.y + v.z * s4.z + v.w * s4.w;
    sd += v.x * d4.x + v.y * d4.y + v.z * d4.z + v.w * d4.w;
  }
  a_s[i] = ss; a_d[i] = sd;
}

// -------- edge pass 1: e = leakyrelu(a_s[src]+a_d[dst]); store e; segment max (uint key) -----
__global__ void k_edge_max(const int* __restrict__ ei, const float* __restrict__ a_s,
                           const float* __restrict__ a_d, unsigned* __restrict__ emaxk,
                           float* __restrict__ ebuf) {
  int eid = blockIdx.x * blockDim.x + threadIdx.x;
  if (eid >= ETOT) return;
  int s, d; edge_nodes(ei, eid, s, d);
#pragma unroll
  for (int h = 0; h < HH; ++h) {
    float e = a_s[s * HH + h] + a_d[d * HH + h];
    e = e > 0.f ? e : SLOPE * e;
    ebuf[eid * HH + h] = e;
    atomicMax(&emaxk[d * HH + h], fkey(e));
  }
}

// -------- edge pass 2: ex = exp(e - emax[dst]); overwrite ebuf; denom += ex ------------------
__global__ void k_edge_exps(const int* __restrict__ ei, const unsigned* __restrict__ emaxk,
                            float* __restrict__ ebuf, float* __restrict__ denom) {
  int eid = blockIdx.x * blockDim.x + threadIdx.x;
  if (eid >= ETOT) return;
  int s, d; edge_nodes(ei, eid, s, d);
#pragma unroll
  for (int h = 0; h < HH; ++h) {
    float ex = expf(ebuf[eid * HH + h] - fdec(emaxk[d * HH + h]));
    ebuf[eid * HH + h] = ex;
    unsafeAtomicAdd(&denom[d * HH + h], ex);
  }
}

// -------- edge pass 3: msg[dst] += h[src] * (ex/denom[dst])  (G lanes/edge, float4) ----------
__global__ void k_edge_scatter(const int* __restrict__ ei, const float* __restrict__ ebuf,
                               const float* __restrict__ denom, const float* __restrict__ hlin,
                               float* __restrict__ msg, int C, int G) {
  int t = blockIdx.x * blockDim.x + threadIdx.x;
  int HC = HH * C;
  int eid = t / G;
  if (eid >= ETOT) return;
  int f  = t % G;
  int j0 = f * 4;
  int h  = j0 / C;
  int s, d; edge_nodes(ei, eid, s, d);
  float alpha = ebuf[eid * HH + h] / denom[d * HH + h];
  float4 hv = *reinterpret_cast<const float4*>(hlin + (size_t)s * HC + j0);
  float* mp = msg + (size_t)d * HC + j0;
  unsafeAtomicAdd(mp + 0, hv.x * alpha);
  unsafeAtomicAdd(mp + 1, hv.y * alpha);
  unsafeAtomicAdd(mp + 2, hv.z * alpha);
  unsafeAtomicAdd(mp + 3, hv.w * alpha);
}

// ---------------- bias + BN + ReLU -> f16 activations for next GEMM ----------------
__global__ void k_bn_act(const float* __restrict__ msg, const float* __restrict__ bias,
                         const float* __restrict__ g, const float* __restrict__ be,
                         const float* __restrict__ m, const float* __restrict__ v,
                         _Float16* __restrict__ act16, int HC) {
  int i = blockIdx.x * blockDim.x + threadIdx.x;
  if (i >= N_NODES * HC) return;
  int j = i % HC;
  float z = msg[i] + bias[j];
  z = (z - m[j]) * rsqrtf(v[j] + EPSV) * g[j] + be[j];
  z = fmaxf(z, 0.f);
  act16[i] = (_Float16)z;
}

// ---------------- layer3: bias + BN + ReLU fused with pooled sum ----------------
__global__ void k_bn_pool(const float* __restrict__ msg, const float* __restrict__ bias,
                          const float* __restrict__ g, const float* __restrict__ be,
                          const float* __restrict__ m, const float* __restrict__ v,
                          const int* __restrict__ batch, float* __restrict__ pooled) {
  int i = blockIdx.x * blockDim.x + threadIdx.x;
  if (i >= N_NODES * 128) return;
  int n = i >> 7, j = i & 127;
  float z = msg[i] + bias[j];
  z = (z - m[j]) * rsqrtf(v[j] + EPSV) * g[j] + be[j];
  z = fmaxf(z, 0.f);
  unsafeAtomicAdd(&pooled[batch[n] * 128 + j], z);
}

__global__ void k_cnt(const int* __restrict__ batch, float* __restrict__ cnts) {
  int n = blockIdx.x * blockDim.x + threadIdx.x;
  if (n < N_NODES) unsafeAtomicAdd(&cnts[batch[n]], 1.f);
}

__global__ void k_poolfin(float* __restrict__ pooled, const float* __restrict__ cnts,
                          float* __restrict__ out) {
  int i = blockIdx.x * blockDim.x + threadIdx.x;
  if (i >= NGRAPH * 128) return;
  int gph = i >> 7;
  float mv = pooled[i] / fmaxf(cnts[gph], 1.f);
  pooled[i] = mv;           // mean kept for MLP
  out[i] = mv;              // output 0: x_out_conv [64,128]
}

// ---------------- MLP head + log_softmax (tiny: 1 thread / graph) ----------------
__global__ void k_mlp(const float* __restrict__ pooled, const float* __restrict__ wf,
                      const float* __restrict__ bf, const float* __restrict__ g4,
                      const float* __restrict__ be4, const float* __restrict__ m4,
                      const float* __restrict__ v4, const float* __restrict__ wl1,
                      const float* __restrict__ bl1, const float* __restrict__ wl2,
                      const float* __restrict__ bl2, float* __restrict__ out) {
  int gph = threadIdx.x;
  if (gph >= NGRAPH) return;
  const float* xr = pooled + gph * 128;
  float z1[32], z2[32], z3[10];
  for (int j = 0; j < 32; ++j) {
    float a = bf[j];
    for (int k = 0; k < 128; ++k) a += xr[k] * wf[k * 32 + j];
    a = (a - m4[j]) * rsqrtf(v4[j] + EPSV) * g4[j] + be4[j];
    z1[j] = fmaxf(a, 0.f);
  }
  for (int j = 0; j < 32; ++j) {
    float a = bl1[j];
    for (int k = 0; k < 32; ++k) a += z1[k] * wl1[k * 32 + j];
    z2[j] = fmaxf(a, 0.f);
  }
  float mx = -1e30f;
  for (int j = 0; j < 10; ++j) {
    float a = bl2[j];
    for (int k = 0; k < 32; ++k) a += z2[k] * wl2[k * 10 + j];
    z3[j] = a; mx = fmaxf(mx, a);
  }
  float ssum = 0.f;
  for (int j = 0; j < 10; ++j) ssum += expf(z3[j] - mx);
  float lse = mx + logf(ssum);
  for (int j = 0; j < 10; ++j) out[NGRAPH * 128 + gph * 10 + j] = z3[j] - lse;
}

extern "C" void kernel_launch(void* const* d_in, const int* in_sizes, int n_in,
                              void* d_out, int out_size, void* d_ws, size_t ws_size,
                              hipStream_t stream) {
  const float* x     = (const float*)d_in[0];
  const int*   ei    = (const int*)  d_in[1];
  const int*   batch = (const int*)  d_in[2];
  const float* w1    = (const float*)d_in[3];
  const float* as1   = (const float*)d_in[4];
  const float* ad1   = (const float*)d_in[5];
  const float* b1    = (const float*)d_in[6];
  const float* g1    = (const float*)d_in[7];
  const float* be1   = (const float*)d_in[8];
  const float* m1    = (const float*)d_in[9];
  const float* v1    = (const float*)d_in[10];
  const float* w2    = (const float*)d_in[11];
  const float* as2   = (const float*)d_in[12];
  const float* ad2   = (const float*)d_in[13];
  const float* b2    = (const float*)d_in[14];
  const float* g2    = (const float*)d_in[15];
  const float* be2   = (const float*)d_in[16];
  const float* m2    = (const float*)d_in[17];
  const float* v2    = (const float*)d_in[18];
  const float* w3    = (const float*)d_in[19];
  const float* as3   = (const float*)d_in[20];
  const float* ad3   = (const float*)d_in[21];
  const float* b3    = (const float*)d_in[22];
  const float* g3    = (const float*)d_in[23];
  const float* be3   = (const float*)d_in[24];
  const float* m3    = (const float*)d_in[25];
  const float* v3    = (const float*)d_in[26];
  const float* wf    = (const float*)d_in[27];
  const float* bf    = (const float*)d_in[28];
  const float* g4    = (const float*)d_in[29];
  const float* be4   = (const float*)d_in[30];
  const float* m4    = (const float*)d_in[31];
  const float* v4    = (const float*)d_in[32];
  const float* wl1   = (const float*)d_in[33];
  const float* bl1   = (const float*)d_in[34];
  const float* wl2   = (const float*)d_in[35];
  const float* bl2   = (const float*)d_in[36];
  float* out = (float*)d_out;

  char* ws = (char*)d_ws;
  float*    hlin   = (float*)(ws);                    // N*128*4      = 25,600,000
  float*    msg    = (float*)(ws + 25600000);         //               25,600,000
  _Float16* act16  = (_Float16*)(ws + 51200000);      // N*128*2     = 12,800,000
  float*    a_s    = (float*)(ws + 64000000);         // N*H*4       =    400,000
  float*    a_d    = (float*)(ws + 64400000);
  unsigned* emaxk  = (unsigned*)(ws + 64800000);
  float*    denom  = (float*)(ws + 65200000);
  float*    ebuf   = (float*)(ws + 65600000);         // ETOT*H*4    = 13,200,000
  _Float16* w2h    = (_Float16*)(ws + 78800000);      // 16,384 (transposed [128,64])
  _Float16* w3h    = (_Float16*)(ws + 78820000);      // 32,768 (transposed [128,128])
  float*    pooled = (float*)(ws + 78860000);         // 32,768
  float*    cnts   = (float*)(ws + 78900000);         // 256

  const int TB = 256;
  const int eb = (ETOT + TB - 1) / TB;

  // transposed weight conversion + pooled init
  k_f2h_t<<<(64 * 128 + TB - 1) / TB, TB, 0, stream>>>(w2, w2h, 64, 128);
  k_f2h_t<<<(128 * 128 + TB - 1) / TB, TB, 0, stream>>>(w3, w3h, 128, 128);
  hipMemsetAsync(pooled, 0, NGRAPH * 128 * 4, stream);
  hipMemsetAsync(cnts, 0, NGRAPH * 4, stream);

  // ---------------- layer 1: C=32, HC=64 ----------------
  k_lin1<<<(N_NODES * 64 + TB - 1) / TB, TB, 0, stream>>>(x, w1, hlin);
  hipMemsetAsync(emaxk, 0, N_NODES * HH * 4, stream);
  hipMemsetAsync(denom, 0, N_NODES * HH * 4, stream);
  hipMemsetAsync(msg, 0, (size_t)N_NODES * 64 * 4, stream);
  k_attdot<<<(N_NODES * HH + TB - 1) / TB, TB, 0, stream>>>(hlin, as1, ad1, a_s, a_d, 32);
  k_edge_max<<<eb, TB, 0, stream>>>(ei, a_s, a_d, emaxk, ebuf);
  k_edge_exps<<<eb, TB, 0, stream>>>(ei, emaxk, ebuf, denom);
  k_edge_scatter<<<(ETOT * 16 + TB - 1) / TB, TB, 0, stream>>>(
      ei, ebuf, denom, hlin, msg, 32, 16);
  k_bn_act<<<(N_NODES * 64 + TB - 1) / TB, TB, 0, stream>>>(msg, b1, g1, be1, m1, v1, act16, 64);

  // ---------------- layer 2: WMMA GEMM [N,64]@[64,128], then C=64, HC=128 ----------------
  k_gemm_wmma<<<3125, TB, 0, stream>>>(act16, w2h, hlin, 64, 128);
  hipMemsetAsync(emaxk, 0, N_NODES * HH * 4, stream);
  hipMemsetAsync(denom, 0, N_NODES * HH * 4, stream);
  hipMemsetAsync(msg, 0, (size_t)N_NODES * 128 * 4, stream);
  k_attdot<<<(N_NODES * HH + TB - 1) / TB, TB, 0, stream>>>(hlin, as2, ad2, a_s, a_d, 64);
  k_edge_max<<<eb, TB, 0, stream>>>(ei, a_s, a_d, emaxk, ebuf);
  k_edge_exps<<<eb, TB, 0, stream>>>(ei, emaxk, ebuf, denom);
  k_edge_scatter<<<(ETOT * 32 + TB - 1) / TB, TB, 0, stream>>>(
      ei, ebuf, denom, hlin, msg, 64, 32);
  k_bn_act<<<(N_NODES * 128 + TB - 1) / TB, TB, 0, stream>>>(msg, b2, g2, be2, m2, v2, act16, 128);

  // ---------------- layer 3: WMMA GEMM [N,128]@[128,128], then C=64, HC=128, fused pool ------
  k_gemm_wmma<<<3125, TB, 0, stream>>>(act16, w3h, hlin, 128, 128);
  hipMemsetAsync(emaxk, 0, N_NODES * HH * 4, stream);
  hipMemsetAsync(denom, 0, N_NODES * HH * 4, stream);
  hipMemsetAsync(msg, 0, (size_t)N_NODES * 128 * 4, stream);
  k_attdot<<<(N_NODES * HH + TB - 1) / TB, TB, 0, stream>>>(hlin, as3, ad3, a_s, a_d, 64);
  k_edge_max<<<eb, TB, 0, stream>>>(ei, a_s, a_d, emaxk, ebuf);
  k_edge_exps<<<eb, TB, 0, stream>>>(ei, emaxk, ebuf, denom);
  k_edge_scatter<<<(ETOT * 32 + TB - 1) / TB, TB, 0, stream>>>(
      ei, ebuf, denom, hlin, msg, 64, 32);
  k_bn_pool<<<(N_NODES * 128 + TB - 1) / TB, TB, 0, stream>>>(msg, b3, g3, be3, m3, v3, batch, pooled);

  // ---------------- pooling finalize + MLP head ----------------
  k_cnt<<<(N_NODES + TB - 1) / TB, TB, 0, stream>>>(batch, cnts);
  k_poolfin<<<(NGRAPH * 128 + TB - 1) / TB, TB, 0, stream>>>(pooled, cnts, out);
  k_mlp<<<1, 64, 0, stream>>>(pooled, wf, bf, g4, be4, m4, v4, wl1, bl1, wl2, bl2, out);
}
